// CoupledGNO_33827162423321
// MI455X (gfx1250) — compile-verified, matrix operand order checked by reading, not run on hardware
//
#include <hip/hip_runtime.h>

#define NN     50000
#define EE     800000
#define INF    3
#define EMB    16
#define EDGEF  6
#define KW     128
#define NCONV  4
#define NTILES (EE / 16)

typedef __attribute__((ext_vector_type(16))) _Float16 v16h;
typedef __attribute__((ext_vector_type(8)))  float    v8f;

// ---------------------------------------------------------------- utility
__global__ void zero_f32(float* __restrict__ p, int n) {
    int i = blockIdx.x * blockDim.x + threadIdx.x;
    if (i < n) p[i] = 0.f;
}

// h0[n,o] = emb_b[o] + sum_j x[n,j] * emb_w[j,o]
__global__ void embed_kernel(const float* __restrict__ x,
                             const float* __restrict__ w,
                             const float* __restrict__ b,
                             float* __restrict__ h) {
    int i = blockIdx.x * blockDim.x + threadIdx.x;   // over NN*EMB
    int n = i >> 4, o = i & 15;
    if (n < NN) {
        float s = b[o];
#pragma unroll
        for (int j = 0; j < INF; ++j) s += x[n * INF + j] * w[j * EMB + o];
        h[i] = s;
    }
}

__global__ void count_kernel(const int* __restrict__ dst, float* __restrict__ cnt) {
    int e = blockIdx.x * blockDim.x + threadIdx.x;
    if (e < EE) atomicAdd(&cnt[dst[e]], 1.0f);
}

// ------------------------------------------------- edge-kernel MLP (WMMA)
// Per 16-edge tile:  act1 = relu(ea@k1+b1)   (VALU, K=6)
//                    act2 = relu(act1@k2+b2) (WMMA f16 16x16x32, K=128)
//                    W    = act2@k3+b3       (WMMA f16 16x16x32, N=256)
// B tiles are loop-invariant per wave -> hoisted into VGPRs once per block.
// W stored f16, transposed per edge: Wt[e][o][i]  (o = out col, i = in row)
__global__ __launch_bounds__(256) void edge_mlp_wmma(
    const float* __restrict__ eattr,
    const float* __restrict__ k1w, const float* __restrict__ k1b,
    const float* __restrict__ k2w, const float* __restrict__ k2b,
    const float* __restrict__ k3w, const float* __restrict__ k3b,
    _Float16* __restrict__ Wt)
{
    __shared__ __align__(32) _Float16 k2s[16384];   // 32 KB, B-swizzled (staging)
    __shared__ __align__(32) _Float16 k3s[32768];   // 64 KB, B-swizzled (staging)
    __shared__ float k1s[EDGEF * KW];
    __shared__ float b1s[KW], b2s[KW], b3s[256];
    __shared__ __align__(32) _Float16 act1[2048];   // 16x128, A-swizzled
    __shared__ __align__(32) _Float16 act2[2048];   // 16x128, A-swizzled
    __shared__ __align__(16) _Float16 wstage[4096]; // 16x256 output staging
    __shared__ float eas[16 * EDGEF];

    const int tid  = threadIdx.x;
    const int lane = tid & 31;
    const int wid  = tid >> 5;           // 8 waves
    const int o    = lane & 15;          // C-matrix column for this lane
    const int mbase = (lane >> 4) * 8;   // C-matrix row base for this lane

    // Preload weights into LDS in WMMA per-lane swizzle:
    //   halves e=0..7  hold K = grp*8 + e        (grp = lane>>4)
    //   halves e=8..15 hold K = 16 + grp*8 + e-8
    for (int p = tid; p < 16384; p += 256) {
        int elem = p & 15, ln = (p >> 4) & 31, nt = (p >> 9) & 7, ks = p >> 12;
        int grp = ln >> 4, col = nt * 16 + (ln & 15);
        int kr = ks * 32 + grp * 8 + (elem & 7) + ((elem >> 3) << 4);
        k2s[p] = (_Float16)k2w[kr * KW + col];
    }
    for (int p = tid; p < 32768; p += 256) {
        int elem = p & 15, ln = (p >> 4) & 31, nt = (p >> 9) & 15, ks = p >> 13;
        int grp = ln >> 4, col = nt * 16 + (ln & 15);
        int kr = ks * 32 + grp * 8 + (elem & 7) + ((elem >> 3) << 4);
        k3s[p] = (_Float16)k3w[kr * 256 + col];
    }
    for (int p = tid; p < EDGEF * KW; p += 256) k1s[p] = k1w[p];
    if (tid < KW) { b1s[tid] = k1b[tid]; b2s[tid] = k2b[tid]; }
    b3s[tid] = k3b[tid];
    __syncthreads();

    const v16h* A1 = (const v16h*)act1;
    const v16h* A2 = (const v16h*)act2;
    const v16h* B2 = (const v16h*)k2s;
    const v16h* B3 = (const v16h*)k3s;

    // Hoist the per-wave B operands into VGPRs (loop-invariant across tiles).
    v16h b2r[4];
#pragma unroll
    for (int kc = 0; kc < 4; ++kc) b2r[kc] = B2[(kc * 8 + wid) * 32 + lane];
    v16h b3r[2][4];
#pragma unroll
    for (int hh = 0; hh < 2; ++hh) {
#pragma unroll
        for (int kc = 0; kc < 4; ++kc)
            b3r[hh][kc] = B3[(kc * 16 + wid + hh * 8) * 32 + lane];
    }

    for (int tile = blockIdx.x; tile < NTILES; tile += gridDim.x) {
        const int e0 = tile * 16;
        if (tid < 16 * EDGEF) {
            eas[tid] = eattr[e0 * EDGEF + tid];
            int nxt = tile + gridDim.x;
            if (nxt < NTILES)
                __builtin_prefetch(&eattr[nxt * 16 * EDGEF + tid], 0, 0);
        }
        __syncthreads();

        // ---- layer 1 (K=6, VALU) directly into A-swizzle layout
        for (int f = tid; f < 2048; f += 256) {
            int m = f >> 7, k = f & 127;
            float s = b1s[k];
#pragma unroll
            for (int j = 0; j < EDGEF; ++j) s += eas[m * EDGEF + j] * k1s[j * KW + k];
            s = fmaxf(s, 0.f);
            int kc = k >> 5, kk = k & 31;
            int grp = (kk >> 3) & 1, elem = (kk & 7) + ((kk >> 4) << 3);
            act1[kc * 512 + (grp * 16 + m) * 16 + elem] = (_Float16)s;
        }
        __syncthreads();

        // ---- layer 2: each wave computes one 16-col N-tile, K=128
        v8f acc = {};
#pragma unroll
        for (int kc = 0; kc < 4; ++kc) {
            v16h a = A1[kc * 32 + lane];
            acc = __builtin_amdgcn_wmma_f32_16x16x32_f16(
                false, a, false, b2r[kc], (short)0, acc, false, false);
        }
        {
            int col = wid * 16 + o;
            float bias = b2s[col];
            int kc2 = col >> 5, kk2 = col & 31;
            int grp2 = (kk2 >> 3) & 1, elem2 = (kk2 & 7) + ((kk2 >> 4) << 3);
#pragma unroll
            for (int r = 0; r < 8; ++r) {
                float v = fmaxf(acc[r] + bias, 0.f);
                act2[kc2 * 512 + (grp2 * 16 + mbase + r) * 16 + elem2] = (_Float16)v;
            }
        }
        __syncthreads();

        // ---- layer 3: load A once, each wave computes two N-tiles of 256 cols
        v16h a2r[4];
#pragma unroll
        for (int kc = 0; kc < 4; ++kc) a2r[kc] = A2[kc * 32 + lane];
#pragma unroll
        for (int hh = 0; hh < 2; ++hh) {
            int nt = wid + hh * 8;   // nt == i (row index of per-edge W)
            v8f acc3 = {};
#pragma unroll
            for (int kc = 0; kc < 4; ++kc) {
                acc3 = __builtin_amdgcn_wmma_f32_16x16x32_f16(
                    false, a2r[kc], false, b3r[hh][kc], (short)0, acc3, false, false);
            }
            float bias = b3s[nt * 16 + o];
#pragma unroll
            for (int r = 0; r < 8; ++r) {
                float v = acc3[r] + bias;
                // transpose to [edge][o][i] for the message kernel
                wstage[(mbase + r) * 256 + o * 16 + nt] = (_Float16)v;
            }
        }
        __syncthreads();

        // coalesced 8 KB copy-out of the tile's weights
        {
            const uint4* s4 = (const uint4*)wstage;
            uint4* g4 = (uint4*)(Wt) + (size_t)tile * 512;
            g4[tid]       = s4[tid];
            g4[tid + 256] = s4[tid + 256];
        }
    }
}

// -------------------------------------------- message pass + scatter-sum
// msg[e,o] = sum_i h[src[e],i] * Wt[e][o][i];  atomic scatter into agg[dst]
__global__ __launch_bounds__(256) void msg_kernel(
    const _Float16* __restrict__ Wt,
    const int* __restrict__ src, const int* __restrict__ dst,
    const float* __restrict__ h, float* __restrict__ agg)
{
    __shared__ __align__(16) _Float16 wl[4096];  // 16 edges x 256
    __shared__ float hl[256];                    // 16 edges x 16 feats
    __shared__ int   sl[16], dl[16];

    const int tid  = threadIdx.x;
    const int tile = blockIdx.x;

    const uint4* g4 = (const uint4*)Wt + (size_t)tile * 512;
    uint4* w4 = (uint4*)wl;
    w4[tid]       = g4[tid];
    w4[tid + 256] = g4[tid + 256];
    if (tid < 16) { sl[tid] = src[tile * 16 + tid]; dl[tid] = dst[tile * 16 + tid]; }
    __syncthreads();
    hl[tid] = h[sl[tid >> 4] * EMB + (tid & 15)];
    __syncthreads();

    const int m = tid >> 4, o = tid & 15;
    const _Float16* wrow = wl + m * 256 + o * 16;  // 32B contiguous per thread
    float s = 0.f;
#pragma unroll
    for (int i = 0; i < 16; ++i) s += hl[m * 16 + i] * (float)wrow[i];
    atomicAdd(&agg[dl[m] * EMB + o], s);
}

// h_out = relu(agg/max(cnt,1) + h_in @ root_w + conv_b)
__global__ __launch_bounds__(256) void combine_kernel(
    const float* __restrict__ agg, const float* __restrict__ cnt,
    const float* __restrict__ rootw, const float* __restrict__ cb,
    const float* __restrict__ hin, float* __restrict__ hout)
{
    __shared__ float rws[256];
    __shared__ float hl[256];
    const int tid = threadIdx.x;
    rws[tid] = rootw[tid];
    const int n = blockIdx.x * 16 + (tid >> 4);
    const int o = tid & 15;
    hl[tid] = (n < NN) ? hin[n * EMB + o] : 0.f;
    __syncthreads();
    if (n < NN) {
        float s = cb[o] + agg[n * EMB + o] / fmaxf(cnt[n], 1.f);
#pragma unroll
        for (int i = 0; i < 16; ++i) s += hl[(tid & ~15) + i] * rws[i * EMB + o];
        hout[n * EMB + o] = fmaxf(s, 0.f);
    }
}

// out[n,j] = inv_b[j] + sum_i h[n,i] * inv_w[i,j]
__global__ void out_kernel(const float* __restrict__ h,
                           const float* __restrict__ invw,
                           const float* __restrict__ invb,
                           float* __restrict__ out) {
    int idx = blockIdx.x * blockDim.x + threadIdx.x;
    if (idx < NN * INF) {
        int n = idx / INF, j = idx % INF;
        float s = invb[j];
#pragma unroll
        for (int i = 0; i < EMB; ++i) s += h[n * EMB + i] * invw[i * INF + j];
        out[idx] = s;
    }
}

// ----------------------------------------------------------------- driver
extern "C" void kernel_launch(void* const* d_in, const int* in_sizes, int n_in,
                              void* d_out, int out_size, void* d_ws, size_t ws_size,
                              hipStream_t stream) {
    (void)in_sizes; (void)n_in; (void)out_size; (void)ws_size;
    const float* x     = (const float*)d_in[0];
    const int*   ei    = (const int*)d_in[1];
    const float* eattr = (const float*)d_in[2];
    const float* emb_w = (const float*)d_in[3];
    const float* emb_b = (const float*)d_in[4];
    const float* k1w   = (const float*)d_in[5];
    const float* k1b   = (const float*)d_in[6];
    const float* k2w   = (const float*)d_in[7];
    const float* k2b   = (const float*)d_in[8];
    const float* k3w   = (const float*)d_in[9];
    const float* k3b   = (const float*)d_in[10];
    const float* rootw = (const float*)d_in[11];
    const float* convb = (const float*)d_in[12];
    const float* invw  = (const float*)d_in[13];
    const float* invb  = (const float*)d_in[14];
    float* out = (float*)d_out;

    const int* src = ei;
    const int* dst = ei + EE;

    char* ws = (char*)d_ws;
    size_t off = 0;
    _Float16* Wt = (_Float16*)(ws + off); off += (size_t)EE * 256 * sizeof(_Float16);
    off = (off + 255) & ~(size_t)255;
    float* hA  = (float*)(ws + off); off += (size_t)NN * EMB * sizeof(float);
    float* hB  = (float*)(ws + off); off += (size_t)NN * EMB * sizeof(float);
    float* agg = (float*)(ws + off); off += (size_t)NN * EMB * sizeof(float);
    float* cnt = (float*)(ws + off); off += (size_t)NN * sizeof(float);

    // degree counts (shared across iterations)
    zero_f32<<<(NN + 255) / 256, 256, 0, stream>>>(cnt, NN);
    count_kernel<<<(EE + 255) / 256, 256, 0, stream>>>(dst, cnt);

    // node embedding
    embed_kernel<<<(NN * EMB + 255) / 256, 256, 0, stream>>>(x, emb_w, emb_b, hA);

    // edge-conditioned weights (WMMA), computed once
    edge_mlp_wmma<<<1024, 256, 0, stream>>>(eattr, k1w, k1b, k2w, k2b, k3w, k3b, Wt);

    float* hin = hA; float* hout = hB;
    for (int it = 0; it < NCONV; ++it) {
        zero_f32<<<(NN * EMB + 255) / 256, 256, 0, stream>>>(agg, NN * EMB);
        msg_kernel<<<NTILES, 256, 0, stream>>>(Wt, src, dst, hin, agg);
        combine_kernel<<<(NN + 15) / 16, 256, 0, stream>>>(agg, cnt, rootw, convb, hin, hout);
        float* t = hin; hin = hout; hout = t;
    }
    // NCONV is even -> final h is back in hA (== hin after swaps)
    out_kernel<<<(NN * INF + 255) / 256, 256, 0, stream>>>(hin, invw, invb, out);
}
